// WindowAttention_16698832847103
// MI455X (gfx1250) — matrix operand order repaired and checked by reading, not compiled, not run on hardware
//
#include <hip/hip_runtime.h>

// CDNA5 / gfx1250 fused Swin window attention, 3-kernel pipeline:
//   prep_weights: f32 -> bf16 weight conversion (once, 128 KB ws)
//   prep_bias:    bias_table[rel_index]+mask baked into padded transposed
//                 tiles bm[widx][h][n][m] (once, 4 MB ws, L2-resident)
//   win_attn:     one workgroup (8 wave32) per window; bf16 WMMA, f32 acc;
//                 all 4 heads processed per phase (3 barriers for attention);
//                 bias enters through the WMMA C operand.

typedef __attribute__((ext_vector_type(16))) __bf16 v16bf;
typedef __attribute__((ext_vector_type(8)))  float  v8f;

#define SEQ   49
#define SP    64      // padded sequence
#define DIMC  128
#define HEADS 4
#define HD    32
#define NW    64

#define WB_ELEMS   (4 * DIMC * DIMC)            // 65536 bf16
#define BM_OFFSET  (WB_ELEMS * 2)               // 128 KB
#define BM_ELEMS   (NW * HEADS * SP * SP)       // 1,048,576 f32

__device__ __forceinline__ unsigned short f2bf(float f) {
  union { __bf16 h; unsigned short s; } u;
  u.h = (__bf16)f;
  return u.s;
}
__device__ __forceinline__ unsigned int f2bf2(float a, float b) {
  union { __bf16 h[2]; unsigned int w; } u;   // -> v_cvt_pk_bf16_f32
  u.h[0] = (__bf16)a; u.h[1] = (__bf16)b;
  return u.w;
}

union BFrag {
  v16bf v;
  unsigned short u[16];
  unsigned int   w[8];
  uint4          q[2];
};
union CFrag {
  v8f    v;
  float4 q[2];
};

// ---- pre-kernel 1: pack the four weight matrices to bf16 ----
__global__ void prep_weights(const float* __restrict__ wq, const float* __restrict__ wk,
                             const float* __restrict__ wv, const float* __restrict__ pw,
                             unsigned short* __restrict__ wb) {
  int i = blockIdx.x * 256 + threadIdx.x;          // pair index
  int m = i * 2;
  if (m >= WB_ELEMS) return;
  const float* src = (m < 16384) ? wq : (m < 32768) ? wk : (m < 49152) ? wv : pw;
  int off = m & 16383;
  float2 p = *reinterpret_cast<const float2*>(src + off);
  *reinterpret_cast<unsigned int*>(wb + m) = f2bf2(p.x, p.y);
}

// ---- pre-kernel 2: padded transposed bias+mask tiles, WMMA-C layout ----
__global__ void prep_bias(const float* __restrict__ mask, const float* __restrict__ bt,
                          const int* __restrict__ ri, float* __restrict__ bm) {
  int t = blockIdx.x * 256 + threadIdx.x;          // [w][h][n][m]
  if (t >= BM_ELEMS) return;
  int m = t & (SP - 1);
  int n = (t >> 6) & (SP - 1);
  int h = (t >> 12) & (HEADS - 1);
  int w = t >> 14;
  float v;
  if (n >= SEQ)      v = -1e30f;                    // padded key col -> exp()=0
  else if (m >= SEQ) v = 0.f;                       // padded query row (discarded)
  else               v = bt[ri[m * SEQ + n] * HEADS + h] + mask[(w * SEQ + m) * SEQ + n];
  bm[t] = v;
}

// ---- main fused kernel ----
__global__ __launch_bounds__(256, 1) void win_attn_kernel(
    const float* __restrict__ x, const unsigned short* __restrict__ wb,
    const float* __restrict__ bm, const float* __restrict__ pb,
    float* __restrict__ out)
{
  __shared__ unsigned short xo[SP][DIMC];       // x (bf16), later attn@V output
  __shared__ unsigned short qs[SP][DIMC];       // Q (pre-scaled), [row][dim]
  __shared__ unsigned short ks[SP][DIMC];       // K, [row][dim]
  __shared__ unsigned short vT[DIMC][SP];       // V transposed: [dim][row]
  __shared__ float          scT[HEADS][SP][SP]; // scores, TRANSPOSED [h][n][m]
  __shared__ unsigned short ab[HEADS][SP][SP];  // probs (bf16), [h][m][n]

  const int tid  = threadIdx.x;
  const int wave = tid >> 5;
  const int lane = tid & 31;
  const int lcol = lane & 15;   // A-row / B-col / D-col index
  const int hf   = lane >> 4;   // lane half select
  const int bwin = blockIdx.x;
  const int widx = bwin & (NW - 1);

  // Warm caches with the packed bf16 weights (128 KB total).
  __builtin_prefetch((const char*)wb + tid * 512, 0, 0);

  // ---- phase 1: stage x -> LDS bf16, float4 loads + b64 stores ----
  for (int i = tid; i < SP * DIMC / 4; i += 256) {
    int r = i >> 5, c = (i & 31) << 2;
    float4 p = make_float4(0.f, 0.f, 0.f, 0.f);
    if (r < SEQ)
      p = *reinterpret_cast<const float4*>(x + (bwin * SEQ + r) * DIMC + c);
    uint2 d;
    d.x = f2bf2(p.x, p.y);
    d.y = f2bf2(p.z, p.w);
    *reinterpret_cast<uint2*>(&xo[r][c]) = d;
  }
  __syncthreads();

  const float scale = 0.17677669529663687f;   // HEAD_DIM^-0.5

  // ---- phase 2: QKV projections.  Wave w owns N-tile w; B hoisted over Mt ----
  {
    const int nt = wave;
    for (int mat = 0; mat < 3; ++mat) {
      const unsigned short* w = wb + mat * DIMC * DIMC;   // bf16 [n][k]
      BFrag B[4];
#pragma unroll
      for (int kk = 0; kk < 4; ++kk) {
        const unsigned short* brow = w + (nt * 16 + lcol) * DIMC + kk * 32 + 8 * hf;
        B[kk].q[0] = *reinterpret_cast<const uint4*>(brow);
        B[kk].q[1] = *reinterpret_cast<const uint4*>(brow + 16);
      }
#pragma unroll
      for (int mt = 0; mt < 4; ++mt) {
        v8f acc = {};
#pragma unroll
        for (int kk = 0; kk < 4; ++kk) {
          BFrag A;   // A[m][k]: row = lcol; K runs {kb..kb+7},{kb+16..kb+23}
          const unsigned short* arow = &xo[mt * 16 + lcol][kk * 32 + 8 * hf];
          A.q[0] = *reinterpret_cast<const uint4*>(arow);
          A.q[1] = *reinterpret_cast<const uint4*>(arow + 16);
          acc = __builtin_amdgcn_wmma_f32_16x16x32_bf16(false, A.v, false, B[kk].v,
                                                        (short)0, acc, false, false);
        }
        const int n  = nt * 16 + lcol;
        const int mb = mt * 16 + 8 * hf;
        if (mat == 0) {
#pragma unroll
          for (int e = 0; e < 8; ++e) qs[mb + e][n] = f2bf(acc[e] * scale);
        } else if (mat == 1) {
#pragma unroll
          for (int e = 0; e < 8; ++e) ks[mb + e][n] = f2bf(acc[e]);
        } else {
          // V transposed: rows contiguous -> packed b32 stores
#pragma unroll
          for (int e = 0; e < 8; e += 2)
            *reinterpret_cast<unsigned int*>(&vT[n][mb + e]) =
                f2bf2(acc[e], acc[e + 1]);
        }
      }
    }
  }
  __syncthreads();

  // ---- phase 3a: scores for ALL heads = Q @ K^T + (bias via WMMA C) ----
  // wave w -> nt = w&3, mt in {w>>2, (w>>2)+2}; loop over heads (B reused/head)
  {
    const int nt  = wave & 3;
    const int mt0 = wave >> 2;
    const int ngl = nt * 16 + lcol;
#pragma unroll
    for (int h = 0; h < HEADS; ++h) {
      BFrag B;   // B[d][n] = K[n][d]
      const unsigned short* brow = &ks[ngl][h * HD + 8 * hf];
      B.q[0] = *reinterpret_cast<const uint4*>(brow);
      B.q[1] = *reinterpret_cast<const uint4*>(brow + 16);
      const float* bmh = bm + ((widx * HEADS + h) * SP + ngl) * SP;  // [n][m] row
#pragma unroll
      for (int i = 0; i < 2; ++i) {
        const int mt = mt0 + 2 * i;
        const int mb = mt * 16 + 8 * hf;
        BFrag A;
        const unsigned short* arow = &qs[mt * 16 + lcol][h * HD + 8 * hf];
        A.q[0] = *reinterpret_cast<const uint4*>(arow);
        A.q[1] = *reinterpret_cast<const uint4*>(arow + 16);
        CFrag C;   // C lane layout: col n, 8 consecutive m -> contiguous in bm
        C.q[0] = *reinterpret_cast<const float4*>(bmh + mb);
        C.q[1] = *reinterpret_cast<const float4*>(bmh + mb + 4);
        v8f acc = __builtin_amdgcn_wmma_f32_16x16x32_bf16(false, A.v, false, B.v,
                                                          (short)0, C.v, false, false);
        *reinterpret_cast<float4*>(&scT[h][ngl][mb]) =
            make_float4(acc[0], acc[1], acc[2], acc[3]);
        *reinterpret_cast<float4*>(&scT[h][ngl][mb + 4]) =
            make_float4(acc[4], acc[5], acc[6], acc[7]);
      }
    }
  }
  __syncthreads();

  // ---- phase 3b: softmax, all 256 threads: thread -> (head, row) ----
  {
    const int h = tid >> 6;
    const int r = tid & (SP - 1);
    if (r < SEQ) {
      float mx = -1e30f;
      for (int n = 0; n < SEQ; ++n) mx = fmaxf(mx, scT[h][n][r]);
      float sum = 0.f;
      float ex[SEQ];
      for (int n = 0; n < SEQ; ++n) {
        float e = __expf(scT[h][n][r] - mx);
        ex[n] = e; sum += e;
      }
      float inv = 1.f / sum;
      for (int n = 0; n < SP; n += 2) {
        float a = (n     < SEQ) ? ex[n] * inv     : 0.f;
        float b = (n + 1 < SEQ) ? ex[n + 1] * inv : 0.f;
        *reinterpret_cast<unsigned int*>(&ab[h][r][n]) = f2bf2(a, b);
      }
    } else {
      for (int n = 0; n < SP; n += 2)
        *reinterpret_cast<unsigned int*>(&ab[h][r][n]) = 0u;
    }
  }
  __syncthreads();

  // ---- phase 3c: out = P @ V for ALL heads; wave -> (mt, nt2), loop heads ----
  {
    const int mt  = wave >> 1;
    const int nt2 = wave & 1;
    const int mb  = mt * 16 + 8 * hf;
#pragma unroll
    for (int h = 0; h < HEADS; ++h) {
      v8f acc = {};
#pragma unroll
      for (int kk = 0; kk < 2; ++kk) {
        BFrag A, B;
        const unsigned short* arow = &ab[h][mt * 16 + lcol][kk * 32 + 8 * hf];
        A.q[0] = *reinterpret_cast<const uint4*>(arow);
        A.q[1] = *reinterpret_cast<const uint4*>(arow + 16);
        // B[k][n] = V[k][h*32+nt2*16+n] = vT[dim][k] -> contiguous in k
        const unsigned short* brow = &vT[h * HD + nt2 * 16 + lcol][kk * 32 + 8 * hf];
        B.q[0] = *reinterpret_cast<const uint4*>(brow);
        B.q[1] = *reinterpret_cast<const uint4*>(brow + 16);
        acc = __builtin_amdgcn_wmma_f32_16x16x32_bf16(false, A.v, false, B.v,
                                                      (short)0, acc, false, false);
      }
      const int n = h * HD + nt2 * 16 + lcol;
#pragma unroll
      for (int e = 0; e < 8; ++e)
        xo[mb + e][n] = f2bf(acc[e]);   // reuse x buffer for concatenated heads
    }
  }
  __syncthreads();

  // ---- phase 4: output projection  Z = O @ Pw^T + b.  B hoisted over Mt ----
  {
    const int nt = wave;
    const unsigned short* w = wb + 3 * DIMC * DIMC;     // proj weights, bf16
    BFrag B[4];
#pragma unroll
    for (int kk = 0; kk < 4; ++kk) {
      const unsigned short* brow = w + (nt * 16 + lcol) * DIMC + kk * 32 + 8 * hf;
      B[kk].q[0] = *reinterpret_cast<const uint4*>(brow);
      B[kk].q[1] = *reinterpret_cast<const uint4*>(brow + 16);
    }
    const int   n    = nt * 16 + lcol;
    const float bias = pb[n];
#pragma unroll
    for (int mt = 0; mt < 4; ++mt) {
      v8f acc = {};
#pragma unroll
      for (int kk = 0; kk < 4; ++kk) {
        BFrag A;
        const unsigned short* arow = &xo[mt * 16 + lcol][kk * 32 + 8 * hf];
        A.q[0] = *reinterpret_cast<const uint4*>(arow);
        A.q[1] = *reinterpret_cast<const uint4*>(arow + 16);
        acc = __builtin_amdgcn_wmma_f32_16x16x32_bf16(false, A.v, false, B[kk].v,
                                                      (short)0, acc, false, false);
      }
#pragma unroll
      for (int e = 0; e < 8; ++e) {
        int m = mt * 16 + 8 * hf + e;
        if (m < SEQ)
          out[(bwin * SEQ + m) * DIMC + n] = acc[e] + bias;
      }
    }
  }
}

extern "C" void kernel_launch(void* const* d_in, const int* in_sizes, int n_in,
                              void* d_out, int out_size, void* d_ws, size_t ws_size,
                              hipStream_t stream) {
  const float* x    = (const float*)d_in[0];
  const float* mask = (const float*)d_in[1];
  const float* wq   = (const float*)d_in[2];
  const float* wk   = (const float*)d_in[3];
  const float* wv   = (const float*)d_in[4];
  const float* pw   = (const float*)d_in[5];
  const float* pb   = (const float*)d_in[6];
  const float* bt   = (const float*)d_in[7];
  const int*   ri   = (const int*)d_in[8];
  (void)in_sizes; (void)n_in; (void)out_size; (void)ws_size;

  unsigned short* wb = (unsigned short*)d_ws;                    // 128 KB bf16 weights
  float*          bm = (float*)((char*)d_ws + BM_OFFSET);        // 4 MB bias tiles

  prep_weights<<<(WB_ELEMS / 2 + 255) / 256, 256, 0, stream>>>(wq, wk, wv, pw, wb);
  prep_bias<<<(BM_ELEMS + 255) / 256, 256, 0, stream>>>(mask, bt, ri, bm);
  win_attn_kernel<<<4096, 256, 0, stream>>>(x, wb, bm, pb, (float*)d_out);
}
